// SelfAttentionPairBias_60146722013683
// MI455X (gfx1250) — matrix operand order repaired and checked by reading, not compiled, not run on hardware
//
#include <hip/hip_runtime.h>
#include <hip/hip_bf16.h>

typedef unsigned short u16;
typedef __attribute__((ext_vector_type(16))) __bf16 v16bf;
typedef __attribute__((ext_vector_type(8)))  float  v8f;

#define Bc 2
#define Nc 512
#define Hc 8
#define HMVc 16
#define HSc 32
#define CINc 16
#define SINc 32
#define CZc 128
#define DQK 160           // 16*8 inner-mv + 32 scalar
#define DV  288           // 16*16 mv blades + 32 scalar
#define QKV_O_MV (3*Hc*HMVc)   // 384
#define QKV_O_S  (3*Hc*HSc)    // 768

__device__ __constant__ int GRADES_c[16] = {0,1,1,1,1,2,2,2,2,2,2,3,3,3,3,4};
// inverse of INNER_IDX = [0,2,3,4,8,9,10,14]
__device__ __constant__ int INNER_POS_c[16] = {0,-1,1,2,3,-1,-1,-1,4,5,6,-1,-1,-1,7,-1};

__device__ __forceinline__ u16 f2bf(float f) {
  unsigned int u = __float_as_uint(f);
  unsigned int r = u + 0x7FFFu + ((u >> 16) & 1u);
  return (u16)(r >> 16);
}

// ---------------------------------------------------------------------------
// K0: QKV equi-linear + RoPE + packing into WMMA-friendly bf16 layouts.
//   qf/kf: (B*H, N, DQK) bf16 row-major (feature contiguous)
//   vt:    (B*H, DV, N)  bf16 (key contiguous -> B-fragment friendly)
// ---------------------------------------------------------------------------
__global__ void qkv_pack(const float* __restrict__ mv, const float* __restrict__ sc,
                         const float* __restrict__ w_mv, const float* __restrict__ w_s2mv,
                         const float* __restrict__ w_s,  const float* __restrict__ w_mv2s,
                         u16* __restrict__ qf, u16* __restrict__ kf, u16* __restrict__ vt) {
  int tok = blockIdx.x;                 // 0..B*N-1
  int b = tok / Nc, n = tok % Nc;
  __shared__ float mv_s[CINc * 16];
  __shared__ float s_s[SINc];
  __shared__ float omv[QKV_O_MV * 16];  // 6144 f32
  __shared__ float os[QKV_O_S];         // 768 f32
  int tid = threadIdx.x;                // 256 threads

  mv_s[tid] = mv[(size_t)tok * 256 + tid];
  if (tid < SINc) s_s[tid] = sc[(size_t)tok * SINc + tid];
  __syncthreads();

  // --- mv outputs: thread -> (x = tid&15, o = (tid>>4) + 16*k) ---
  int x = tid & 15;
  int ob = tid >> 4;
  int g = GRADES_c[x];
  #pragma unroll 4
  for (int k = 0; k < 24; ++k) {
    int o = ob + 16 * k;
    const float* wr = w_mv + ((size_t)g * QKV_O_MV + o) * 16;
    float acc = 0.f;
    #pragma unroll
    for (int i = 0; i < 16; ++i) acc += wr[i] * mv_s[i * 16 + x];
    if (x == 0) {
      const float* w2 = w_s2mv + (size_t)o * SINc;
      #pragma unroll
      for (int s2 = 0; s2 < SINc; ++s2) acc += s_s[s2] * w2[s2];
    }
    omv[o * 16 + x] = acc;
  }
  // --- scalar outputs: o = tid + 256*k ---
  #pragma unroll
  for (int k = 0; k < 3; ++k) {
    int o = tid + 256 * k;
    const float* wr = w_s + (size_t)o * SINc;
    float acc = 0.f;
    #pragma unroll
    for (int s2 = 0; s2 < SINc; ++s2) acc += s_s[s2] * wr[s2];
    const float* w2 = w_mv2s + (size_t)o * CINc;
    #pragma unroll
    for (int i = 0; i < CINc; ++i) acc += mv_s[i * 16] * w2[i];
    os[o] = acc;
  }
  __syncthreads();

  // --- pack mv part ---
  #pragma unroll 4
  for (int k = 0; k < 24; ++k) {
    int idx = tid + 256 * k;            // = o*16 + x2
    int o = idx >> 4, x2 = idx & 15;
    int q3 = o / (Hc * HMVc);
    int rem = o - q3 * (Hc * HMVc);
    int h = rem >> 4, c = rem & 15;     // HMV == 16
    float val = omv[idx];
    if (q3 == 2) {
      vt[((size_t)(b * Hc + h) * DV + (c * 16 + x2)) * Nc + n] = f2bf(val);
    } else {
      int ip = INNER_POS_c[x2];
      if (ip >= 0) {
        u16* dst = (q3 == 0) ? qf : kf;
        dst[((size_t)(b * Hc + h) * Nc + n) * DQK + c * 8 + ip] = f2bf(val);
      }
    }
  }
  // --- pack scalar part with RoPE on q/k ---
  #pragma unroll
  for (int k = 0; k < 3; ++k) {
    int o = tid + 256 * k;
    int q3 = o / (Hc * HSc);
    int rem = o - q3 * (Hc * HSc);
    int h = rem >> 5, cs = rem & 31;    // HS == 32
    float val = os[o];
    if (q3 == 2) {
      vt[((size_t)(b * Hc + h) * DV + (256 + cs)) * Nc + n] = f2bf(val);
    } else {
      int i = cs & 15;
      float inv = exp2f(-0.75f * (float)i);   // 4096^(-i/16) = 2^(-12i/16)
      float ang = (float)n * inv;
      float sn, cn; __sincosf(ang, &sn, &cn);
      float partner = os[o ^ 16];             // other rope half (base mult of 32)
      float outv = (cs < 16) ? (val * cn - partner * sn)
                             : (partner * sn + val * cn);
      u16* dst = (q3 == 0) ? qf : kf;
      dst[((size_t)(b * Hc + h) * Nc + n) * DQK + 128 + cs] = f2bf(outv);
    }
  }
}

// ---------------------------------------------------------------------------
// K1: LayerNorm over CZ=128 + projection to 8 heads. One wave per (b,n,m).
// Streams the 256 MB pair tensor exactly once (HBM-bound kernel).
// global_prefetch_b8 pulls the row 64 waves (8 blocks) ahead while the
// cross-lane reductions run. Writes bias (B*H, N, N) f32.
// ---------------------------------------------------------------------------
__global__ void pair_bias(const float* __restrict__ ps, const float* __restrict__ gamma,
                          const float* __restrict__ beta, const float* __restrict__ wb,
                          float* __restrict__ bias) {
  int wid = blockIdx.x * 8 + (threadIdx.x >> 5);   // 0..B*N*N-1
  int lane = threadIdx.x & 31;
  int b = wid / (Nc * Nc);
  int r = wid - b * (Nc * Nc);
  int n = r / Nc, m = r - n * Nc;

  const float* row = ps + (size_t)wid * CZc + lane * 4;
  // prefetch far ahead (the row the wave 8 blocks later will consume)
  __builtin_prefetch(row + (size_t)64 * CZc, 0, 0);  // -> global_prefetch_b8
  const float4 v = *(const float4*)row;
  float s  = v.x + v.y + v.z + v.w;
  float ss = v.x * v.x + v.y * v.y + v.z * v.z + v.w * v.w;
  #pragma unroll
  for (int off = 16; off > 0; off >>= 1) {
    s  += __shfl_xor(s,  off, 32);
    ss += __shfl_xor(ss, off, 32);
  }
  float mu = s * (1.0f / 128.0f);
  float var = ss * (1.0f / 128.0f) - mu * mu;
  float rs = rsqrtf(var + 1e-5f);

  int z0 = lane * 4;
  float z[4];
  z[0] = (v.x - mu) * rs * gamma[z0 + 0] + beta[z0 + 0];
  z[1] = (v.y - mu) * rs * gamma[z0 + 1] + beta[z0 + 1];
  z[2] = (v.z - mu) * rs * gamma[z0 + 2] + beta[z0 + 2];
  z[3] = (v.w - mu) * rs * gamma[z0 + 3] + beta[z0 + 3];

  float outv = 0.f;
  #pragma unroll
  for (int h = 0; h < Hc; ++h) {
    float t = 0.f;
    #pragma unroll
    for (int j = 0; j < 4; ++j) t += z[j] * wb[(size_t)(z0 + j) * Hc + h];
    #pragma unroll
    for (int off = 16; off > 0; off >>= 1) t += __shfl_xor(t, off, 32);
    if (lane == h) outv = t;
  }
  if (lane < Hc)
    bias[((size_t)(b * Hc + lane) * Nc + n) * Nc + m] = outv;
}

// ---- WMMA fragment loaders (ISA 7.12.2 layouts) ----
__device__ __forceinline__ v16bf load_fragA(const u16* rowbase, int hi) {
  // A 16x32 bf16: lane row = l&15; j<8 -> K=8*hi+j ; j>=8 -> K=16+8*hi+(j-8)
  v16bf a;
  uint4* ap = reinterpret_cast<uint4*>(&a);
  ap[0] = *(const uint4*)(rowbase + 8 * hi);
  ap[1] = *(const uint4*)(rowbase + 16 + 8 * hi);
  return a;
}
__device__ __forceinline__ v16bf load_fragB(const u16* colbase, int hi) {
  // B 32x16 bf16: lane col = l&15; j<8 -> K=8*hi+j ; j>=8 -> K=16+8*hi+(j-8)
  v16bf bm;
  uint4* bp = reinterpret_cast<uint4*>(&bm);
  bp[0] = *(const uint4*)(colbase + 8 * hi);
  bp[1] = *(const uint4*)(colbase + 16 + 8 * hi);
  return bm;
}

// ---------------------------------------------------------------------------
// K2a: logits = (q_feat . k_feat) * scale + bias, in place over bias buffer.
// One wave per 16x16 tile, K = 160 -> 5 x v_wmma_f32_16x16x32_bf16.
// qf/kf tiles live in L2 (2.6 MB each), so per-tile re-reads are L2 hits.
// ---------------------------------------------------------------------------
__global__ void logits_k(const u16* __restrict__ qf, const u16* __restrict__ kf,
                         float* __restrict__ lg) {
  int wid = blockIdx.x * 8 + (threadIdx.x >> 5);
  int lane = threadIdx.x & 31;
  int bh = wid >> 10;                    // / (32*32)
  int qt = (wid >> 5) & 31;
  int kt = wid & 31;
  int r = lane & 15, hi = lane >> 4;

  const u16* qbase = qf + ((size_t)bh * Nc + qt * 16 + r) * DQK;
  const u16* kbase = kf + ((size_t)bh * Nc + kt * 16 + r) * DQK;

  v8f c = {};
  #pragma unroll
  for (int kc = 0; kc < 5; ++kc) {
    v16bf a  = load_fragA(qbase + kc * 32, hi);
    v16bf bm = load_fragB(kbase + kc * 32, hi);
    c = __builtin_amdgcn_wmma_f32_16x16x32_bf16(false, a, false, bm,
                                                (short)0, c, false, false);
  }
  const float scale = 0.07905694150420949f;   // (160)^-0.5
  float* base = lg + ((size_t)bh * Nc + qt * 16) * Nc + kt * 16;
  #pragma unroll
  for (int rr = 0; rr < 8; ++rr) {
    float* p = base + (size_t)(rr + 8 * hi) * Nc + r;   // row q, col = lane&15
    *p = c[rr] * scale + *p;                            // add pair bias in place
  }
}

// ---------------------------------------------------------------------------
// K2b: row softmax over 512 keys, output bf16 attention weights.
// ---------------------------------------------------------------------------
__global__ void softmax_k(const float* __restrict__ lg, u16* __restrict__ attn) {
  int row = blockIdx.x * 8 + (threadIdx.x >> 5);   // 0..B*H*N-1
  int lane = threadIdx.x & 31;
  const float* p = lg + (size_t)row * Nc;
  float v[16];
  float mx = -1e30f;
  #pragma unroll
  for (int k = 0; k < 16; ++k) { v[k] = p[k * 32 + lane]; mx = fmaxf(mx, v[k]); }
  #pragma unroll
  for (int off = 16; off > 0; off >>= 1) mx = fmaxf(mx, __shfl_xor(mx, off, 32));
  float sum = 0.f;
  #pragma unroll
  for (int k = 0; k < 16; ++k) { v[k] = __expf(v[k] - mx); sum += v[k]; }
  #pragma unroll
  for (int off = 16; off > 0; off >>= 1) sum += __shfl_xor(sum, off, 32);
  float inv = 1.0f / sum;
  #pragma unroll
  for (int k = 0; k < 16; ++k) attn[(size_t)row * Nc + k * 32 + lane] = f2bf(v[k] * inv);
}

// ---------------------------------------------------------------------------
// K3: H = attn @ V.  Per (b,h): (512x512) @ (512x288), K = 512 -> 16 WMMA steps.
// A from attn bf16 (key-contiguous rows); B from vt (B*H, DV, N) key-contiguous.
// attn (8 MB) + vt (4.7 MB) stay L2-resident across tile re-reads.
// Output hbuf (B*H, N, DV) f32.
// ---------------------------------------------------------------------------
__global__ void av_k(const u16* __restrict__ attn, const u16* __restrict__ vt,
                     float* __restrict__ hbuf) {
  int wid = blockIdx.x * 8 + (threadIdx.x >> 5);   // bh(16) * mt(32) * nt(18)
  int lane = threadIdx.x & 31;
  int bh = wid / (32 * 18);
  int r2 = wid - bh * (32 * 18);
  int mt = r2 / 18, nt = r2 - mt * 18;
  int r = lane & 15, hi = lane >> 4;

  const u16* abase = attn + ((size_t)bh * Nc + mt * 16 + r) * Nc;
  const u16* bbase = vt   + ((size_t)bh * DV + nt * 16 + r) * Nc;

  v8f c = {};
  #pragma unroll 4
  for (int kc = 0; kc < 16; ++kc) {
    v16bf a  = load_fragA(abase + kc * 32, hi);
    v16bf bm = load_fragB(bbase + kc * 32, hi);
    c = __builtin_amdgcn_wmma_f32_16x16x32_bf16(false, a, false, bm,
                                                (short)0, c, false, false);
  }
  #pragma unroll
  for (int rr = 0; rr < 8; ++rr)
    hbuf[((size_t)bh * Nc + mt * 16 + rr + 8 * hi) * DV + nt * 16 + r] = c[rr];
}

// ---------------------------------------------------------------------------
// K4: output equi-linear. One block per token.
// d_out = out_mv (B,N,16,16) f32  ||  out_s (B,N,32) f32
// ---------------------------------------------------------------------------
__global__ void out_k(const float* __restrict__ hbuf,
                      const float* __restrict__ w_mv, const float* __restrict__ w_s2mv,
                      const float* __restrict__ w_s,  const float* __restrict__ w_mv2s,
                      float* __restrict__ out) {
  int tok = blockIdx.x;
  int b = tok / Nc, n = tok % Nc;
  __shared__ float hmv[128 * 16];   // (hc, x)
  __shared__ float hs[256];
  int tid = threadIdx.x;            // 256

  #pragma unroll
  for (int k = 0; k < 8; ++k) {
    int idx = tid + 256 * k;        // hc*16 + x, hc = h*16 + c
    int hc = idx >> 4, x2 = idx & 15;
    int h = hc >> 4, c = hc & 15;
    hmv[idx] = hbuf[((size_t)(b * Hc + h) * Nc + n) * DV + c * 16 + x2];
  }
  {
    int h = tid >> 5, cs = tid & 31;
    hs[tid] = hbuf[((size_t)(b * Hc + h) * Nc + n) * DV + 256 + cs];
  }
  __syncthreads();

  int o = tid >> 4, x = tid & 15;
  int g = GRADES_c[x];
  const float* wr = w_mv + ((size_t)g * 16 + o) * 128;
  float acc = 0.f;
  #pragma unroll 8
  for (int hc = 0; hc < 128; ++hc) acc += wr[hc] * hmv[hc * 16 + x];
  if (x == 0) {
    const float* w2 = w_s2mv + (size_t)o * 256;
    #pragma unroll 8
    for (int j = 0; j < 256; ++j) acc += hs[j] * w2[j];
  }
  out[(size_t)tok * 256 + tid] = acc;

  if (tid < 32) {
    const float* wr2 = w_s + (size_t)tid * 256;
    float a2 = 0.f;
    #pragma unroll 8
    for (int j = 0; j < 256; ++j) a2 += hs[j] * wr2[j];
    const float* wr3 = w_mv2s + (size_t)tid * 128;
    #pragma unroll 8
    for (int hc = 0; hc < 128; ++hc) a2 += hmv[hc * 16] * wr3[hc];
    out[(size_t)Bc * Nc * 256 + (size_t)tok * 32 + tid] = a2;
  }
}

// ---------------------------------------------------------------------------
extern "C" void kernel_launch(void* const* d_in, const int* in_sizes, int n_in,
                              void* d_out, int out_size, void* d_ws, size_t ws_size,
                              hipStream_t stream) {
  const float* mv       = (const float*)d_in[0];
  const float* sc       = (const float*)d_in[1];
  const float* ps       = (const float*)d_in[2];
  const float* w_qkv_mv   = (const float*)d_in[3];
  const float* w_qkv_s2mv = (const float*)d_in[4];
  const float* w_qkv_s    = (const float*)d_in[5];
  const float* w_qkv_mv2s = (const float*)d_in[6];
  const float* ln_gamma   = (const float*)d_in[7];
  const float* ln_beta    = (const float*)d_in[8];
  const float* w_bias     = (const float*)d_in[9];
  const float* w_out_mv   = (const float*)d_in[10];
  const float* w_out_s2mv = (const float*)d_in[11];
  const float* w_out_s    = (const float*)d_in[12];
  const float* w_out_mv2s = (const float*)d_in[13];
  float* out = (float*)d_out;

  char* ws = (char*)d_ws;
  size_t off = 0;
  u16* qf = (u16*)(ws + off); off += (size_t)Bc * Hc * Nc * DQK * 2;          // 2.6 MB
  u16* kf = (u16*)(ws + off); off += (size_t)Bc * Hc * Nc * DQK * 2;          // 2.6 MB
  u16* vt = (u16*)(ws + off); off += (size_t)Bc * Hc * DV * Nc * 2;           // 4.7 MB
  float* biasbuf = (float*)(ws + off); off += (size_t)Bc * Hc * Nc * Nc * 4;  // 16.8 MB
  u16* attn = (u16*)(ws + off); off += (size_t)Bc * Hc * Nc * Nc * 2;         // 8.4 MB
  float* hbuf = (float*)(ws + off); off += (size_t)Bc * Hc * Nc * DV * 4;     // 9.4 MB

  qkv_pack<<<Bc * Nc, 256, 0, stream>>>(mv, sc, w_qkv_mv, w_qkv_s2mv, w_qkv_s,
                                        w_qkv_mv2s, qf, kf, vt);
  pair_bias<<<(Bc * Nc * Nc) / 8, 256, 0, stream>>>(ps, ln_gamma, ln_beta, w_bias, biasbuf);
  logits_k<<<(Bc * Hc * 32 * 32) / 8, 256, 0, stream>>>(qf, kf, biasbuf);
  softmax_k<<<(Bc * Hc * Nc) / 8, 256, 0, stream>>>(biasbuf, attn);
  av_k<<<(Bc * Hc * 32 * 18) / 8, 256, 0, stream>>>(attn, vt, hbuf);
  out_k<<<Bc * Nc, 256, 0, stream>>>(hbuf, w_out_mv, w_out_s2mv, w_out_s, w_out_mv2s, out);
}
